// SelfAttentionLanguageModel_26542897889537
// MI455X (gfx1250) — matrix operand order repaired and checked by reading, not compiled, not run on hardware
//
#include <hip/hip_runtime.h>

// ---------------- problem constants ----------------
#define VOCAB   32000
#define NEMB    1024
#define HSZ     64
#define BATCH   4
#define TLEN    2048
#define ROWS    (BATCH * TLEN)          // 8192

// ---------------- WMMA fragment types ----------------
typedef __attribute__((ext_vector_type(16))) __bf16     v16bf;
typedef __attribute__((ext_vector_type(8)))  float      v8f;
typedef __attribute__((ext_vector_type(4))) unsigned    u32x4;

union BfFrag {            // 32 bytes: 16 bf16 elements = 8 VGPRs/lane
    u32x4 u[2];
    v16bf v;
};

__device__ __forceinline__ unsigned short f2bf(float f) {
    // round-to-nearest-even f32 -> bf16 (maps +-inf correctly, no NaN inputs here)
    unsigned int u = __float_as_uint(f);
    unsigned int r = u + 0x7FFFu + ((u >> 16) & 1u);
    return (unsigned short)(r >> 16);
}
__device__ __forceinline__ float bf2f(unsigned short h) {
    return __uint_as_float(((unsigned int)h) << 16);
}

__device__ __forceinline__ v8f wmma_bf16(const BfFrag& a, const BfFrag& b, v8f c) {
    return __builtin_amdgcn_wmma_f32_16x16x32_bf16(false, a.v, false, b.v,
                                                   (short)0, c, false, false);
}

// A fragment (16x32 bf16, MxK): lane<16: elems0-7=K0..7, elems8-15=K16..23
//                               lane>=16: elems0-7=K8..15, elems8-15=K24..31
// rowbase = &src[(m0 + lane%16) * ld], koff = 32*kstep
__device__ __forceinline__ BfFrag load_a(const unsigned short* rowbase, int koff, int hi) {
    BfFrag f;
    const unsigned short* p = rowbase + koff + (hi ? 8 : 0);
    f.u[0] = *(const u32x4*)(p);
    f.u[1] = *(const u32x4*)(p + 16);
    return f;
}
// B fragment (32x16 bf16, KxN): lane<16: elems0-7=K0..7, elems8-15=K16..23; lane>=16: elems0-7=K8..15, elems8-15=K24..31
// colbase = &src[(n0 + lane%16) * ld] with src stored [N][K] (K contiguous)
__device__ __forceinline__ BfFrag load_b(const unsigned short* colbase, int koff, int hi) {
    BfFrag f;
    const unsigned short* p = colbase + koff + (hi ? 8 : 0);
    f.u[0] = *(const u32x4*)(p);
    f.u[1] = *(const u32x4*)(p + 16);
    return f;
}

// ---------------- 1) embedding gather -> bf16 x [8192][1024] ----------------
__global__ void gather_cvt_kernel(const int* __restrict__ idx,
                                  const float* __restrict__ emb,
                                  unsigned short* __restrict__ xb) {
    const int row = blockIdx.x;
    const float* er = emb + (size_t)idx[row] * NEMB;
    unsigned short* xr = xb + (size_t)row * NEMB;
    for (int i = threadIdx.x; i < NEMB; i += 256) xr[i] = f2bf(er[i]);
}

// ---------------- 2a) Wq|Wk|Wv -> bf16 transposed [192][1024] ----------------
__global__ void cvt_wqkv_kernel(const float* __restrict__ Wq,
                                const float* __restrict__ Wk,
                                const float* __restrict__ Wv,
                                unsigned short* __restrict__ wt) {
    const int n = blockIdx.x;                  // 0..191
    const float* W = (n < 64) ? Wq : (n < 128) ? Wk : Wv;
    const int col = n & 63;
    unsigned short* dst = wt + (size_t)n * NEMB;
    for (int c = threadIdx.x; c < NEMB; c += 256)
        dst[c] = f2bf(W[(size_t)c * HSZ + col]);
}

// ---------------- 2b) QKV GEMM on WMMA: [8192,1024] x [1024,192] -------------
// wave: one 16-row M tile x one 16-col N tile, K=1024 -> 32 wmma k-steps.
// q,k written row-major bf16 [row][h]; v written transposed bf16 vt[b][h][s].
__global__ void qkv_wmma_kernel(const unsigned short* __restrict__ xb,
                                const unsigned short* __restrict__ wt,
                                unsigned short* __restrict__ qb,
                                unsigned short* __restrict__ kb,
                                unsigned short* __restrict__ vt) {
    const int lane = threadIdx.x & 31, wave = threadIdx.x >> 5;
    const int l15 = lane & 15, hi = lane >> 4;
    const int m0 = (blockIdx.y * 8 + wave) * 16;
    const int n0 = blockIdx.x * 16;

    const unsigned short* arow = xb + (size_t)(m0 + l15) * NEMB;
    const unsigned short* brow = wt + (size_t)(n0 + l15) * NEMB;

    v8f c = {};
    for (int kc = 0; kc < NEMB / 32; ++kc) {
        __builtin_prefetch(arow + (kc + 1) * 32, 0, 0);   // global_prefetch of next A chunk
        BfFrag a = load_a(arow, kc * 32, hi);
        BfFrag b = load_b(brow, kc * 32, hi);
        c = wmma_bf16(a, b, c);
    }

    const int which = n0 >> 6;                 // uniform per block: 0:q 1:k 2:v
    const int h = (n0 & 63) + l15;
    #pragma unroll
    for (int r = 0; r < 8; ++r) {
        const int row = m0 + r + hi * 8;
        const unsigned short val = f2bf(c[r]);
        if (which == 0)      qb[(size_t)row * HSZ + h] = val;
        else if (which == 1) kb[(size_t)row * HSZ + h] = val;
        else {
            const int bidx = row >> 11, s = row & (TLEN - 1);
            vt[((size_t)bidx * HSZ + h) * TLEN + s] = val;
        }
    }
}

// ---------------- 3) scores on WMMA: S = Q K^T * 1/32, causal mask -----------
// wave: 16 t-rows x 16 s-cols, K=64 (2 k-steps). Writes bf16 scores [b][t][s].
__global__ void scores_wmma_kernel(const unsigned short* __restrict__ qb,
                                   const unsigned short* __restrict__ kb,
                                   unsigned short* __restrict__ scp) {
    const int lane = threadIdx.x & 31, wave = threadIdx.x >> 5;
    const int l15 = lane & 15, hi = lane >> 4;
    const int b  = blockIdx.z;
    const int t0 = blockIdx.y * 16;
    const int s0 = blockIdx.x * 128 + wave * 16;

    const unsigned short* arow = qb + (size_t)(b * TLEN + t0 + l15) * HSZ;
    const unsigned short* brow = kb + (size_t)(b * TLEN + s0 + l15) * HSZ;

    v8f c = {};
    c = wmma_bf16(load_a(arow, 0, hi),  load_b(brow, 0, hi),  c);
    c = wmma_bf16(load_a(arow, 32, hi), load_b(brow, 32, hi), c);

    const int s = s0 + l15;
    #pragma unroll
    for (int r = 0; r < 8; ++r) {
        const int t = t0 + r + hi * 8;
        float v = c[r] * 0.03125f;             // C**-0.5 = 1/32 (faithful)
        if (s > t) v = -INFINITY;              // causal mask; f2bf(-inf)=0xFF80
        scp[((size_t)b * TLEN + t) * TLEN + s] = f2bf(v);
    }
}

// ---------------- 4) row softmax, in-place bf16, writes normalized P ---------
__global__ void softmax_kernel(unsigned short* __restrict__ scp) {
    const int row = blockIdx.x;                // global row = b*T + t
    const int tid = threadIdx.x;
    unsigned short* base = scp + (size_t)row * TLEN;
    __shared__ float red[256];

    float vals[8];
    float lmax = -INFINITY;
    #pragma unroll
    for (int j = 0; j < 8; ++j) {
        vals[j] = bf2f(base[tid + j * 256]);
        lmax = fmaxf(lmax, vals[j]);
    }
    red[tid] = lmax; __syncthreads();
    for (int off = 128; off > 0; off >>= 1) {
        if (tid < off) red[tid] = fmaxf(red[tid], red[tid + off]);
        __syncthreads();
    }
    const float m = red[0]; __syncthreads();

    float lsum = 0.f;
    #pragma unroll
    for (int j = 0; j < 8; ++j) {
        vals[j] = __expf(vals[j] - m);         // exp(-inf - m) = 0 for masked
        lsum += vals[j];
    }
    red[tid] = lsum; __syncthreads();
    for (int off = 128; off > 0; off >>= 1) {
        if (tid < off) red[tid] += red[tid + off];
        __syncthreads();
    }
    const float inv = 1.f / red[0];
    #pragma unroll
    for (int j = 0; j < 8; ++j)
        base[tid + j * 256] = f2bf(vals[j] * inv);
}

// ---------------- 5) P @ V on WMMA -> out bf16 [8192][64] --------------------
// wave: 16 t-rows x 64 h-cols (4 C tiles), causal-truncated K loop over s.
__global__ void pv_wmma_kernel(const unsigned short* __restrict__ pb,
                               const unsigned short* __restrict__ vt,
                               unsigned short* __restrict__ outb) {
    const int lane = threadIdx.x & 31, wave = threadIdx.x >> 5;
    const int l15 = lane & 15, hi = lane >> 4;
    const int mtile = blockIdx.x * 8 + wave;   // 0..511
    const int b  = mtile >> 7;
    const int t0 = (mtile & 127) * 16;

    const unsigned short* arow  = pb + ((size_t)b * TLEN + t0 + l15) * TLEN;
    const unsigned short* vbase = vt + (size_t)b * HSZ * TLEN + (size_t)l15 * TLEN;

    v8f c0 = {}, c1 = {}, c2 = {}, c3 = {};
    const int nk = (t0 + 47) >> 5;             // k-steps needed (P is 0 past t)
    for (int kc = 0; kc < nk; ++kc) {
        BfFrag a = load_a(arow, kc * 32, hi);
        c0 = wmma_bf16(a, load_b(vbase + 0 * 16 * TLEN, kc * 32, hi), c0);
        c1 = wmma_bf16(a, load_b(vbase + 1 * 16 * TLEN, kc * 32, hi), c1);
        c2 = wmma_bf16(a, load_b(vbase + 2 * 16 * TLEN, kc * 32, hi), c2);
        c3 = wmma_bf16(a, load_b(vbase + 3 * 16 * TLEN, kc * 32, hi), c3);
    }
    #pragma unroll
    for (int r = 0; r < 8; ++r) {
        const size_t row = (size_t)b * TLEN + t0 + r + hi * 8;
        outb[row * HSZ +  0 + l15] = f2bf(c0[r]);
        outb[row * HSZ + 16 + l15] = f2bf(c1[r]);
        outb[row * HSZ + 32 + l15] = f2bf(c2[r]);
        outb[row * HSZ + 48 + l15] = f2bf(c3[r]);
    }
}

// ---------------- 6a) W_lm[:64] transpose+convert -> bf16 [N][K] -------------
__global__ void cvt_wlm_kernel(const float* __restrict__ Wlm,
                               unsigned short* __restrict__ wlmt) {
    int n = blockIdx.x * 256 + threadIdx.x;    // 125*256 = 32000 exact
    unsigned short* dst = wlmt + (size_t)n * HSZ;
    #pragma unroll 8
    for (int kk = 0; kk < HSZ; ++kk)
        dst[kk] = f2bf(Wlm[(size_t)kk * VOCAB + n]);
}

// ---------------- 6b) logits GEMM: [8192,64] x [64,32000] + bias -------------
// All 8 waves in a block share the same A tile -> stage it once per block into
// LDS with async global->LDS DMA (ASYNCcnt path), then feed WMMA from LDS.
__global__ void logits_wmma_kernel(const unsigned short* __restrict__ outb,
                                   const unsigned short* __restrict__ wlmt,
                                   const float* __restrict__ b_lm,
                                   float* __restrict__ logits) {
    __shared__ unsigned short shA[16 * HSZ];   // 16x64 bf16 A tile, 2 KB
    const int tid  = threadIdx.x;
    const int lane = tid & 31, wave = tid >> 5;
    const int l15 = lane & 15, hi = lane >> 4;
    const int m0 = blockIdx.y * 16;
    const int n0 = blockIdx.x * 256 + wave * 32;

    // --- async-stage the shared A tile: 128 lanes x 16B = 2 KB ---
    if (tid < 128) {
        const int row = tid >> 3, seg = tid & 7;           // 8 x 16B segs per row
        const unsigned long long gaddr =
            (unsigned long long)(const void*)(outb + (size_t)(m0 + row) * HSZ + seg * 8);
        const unsigned ldsoff =
            (unsigned)(unsigned long long)(const void*)shA + (unsigned)(row * 128 + seg * 16);
        asm volatile("global_load_async_to_lds_b128 %0, %1, off"
                     :: "v"(ldsoff), "v"(gaddr) : "memory");
    }
    asm volatile("s_wait_asynccnt 0x0" ::: "memory");      // issuing waves drain DMA
    __syncthreads();                                       // publish LDS tile to all waves

    const unsigned short* arow = shA + l15 * HSZ;          // ds_load_b128 fragment reads
    BfFrag a0 = load_a(arow, 0, hi);
    BfFrag a1 = load_a(arow, 32, hi);

    v8f c0 = {}, c1 = {};
    {
        const unsigned short* brow = wlmt + (size_t)(n0 + l15) * HSZ;
        c0 = wmma_bf16(a0, load_b(brow, 0, hi),  c0);
        c0 = wmma_bf16(a1, load_b(brow, 32, hi), c0);
    }
    {
        const unsigned short* brow = wlmt + (size_t)(n0 + 16 + l15) * HSZ;
        c1 = wmma_bf16(a0, load_b(brow, 0, hi),  c1);
        c1 = wmma_bf16(a1, load_b(brow, 32, hi), c1);
    }

    const int col0 = n0 + l15;
    const float bias0 = b_lm[col0];
    const float bias1 = b_lm[col0 + 16];
    #pragma unroll
    for (int r = 0; r < 8; ++r) {
        const size_t row = (size_t)(m0 + r + hi * 8);
        logits[row * VOCAB + col0]      = c0[r] + bias0;
        logits[row * VOCAB + col0 + 16] = c1[r] + bias1;
    }
}

// ---------------- host side ----------------
extern "C" void kernel_launch(void* const* d_in, const int* in_sizes, int n_in,
                              void* d_out, int out_size, void* d_ws, size_t ws_size,
                              hipStream_t stream) {
    const int*   idx  = (const int*)  d_in[0];
    const float* emb  = (const float*)d_in[1];
    const float* Wq   = (const float*)d_in[2];
    const float* Wk   = (const float*)d_in[3];
    const float* Wv   = (const float*)d_in[4];
    const float* Wlm  = (const float*)d_in[5];
    const float* b_lm = (const float*)d_in[6];
    float* logits = (float*)d_out;

    char* ws = (char*)d_ws;
    const size_t OFF_XB   = 0;                              // 16 MB
    const size_t OFF_WQKV = (size_t)16 << 20;               // 384 KB (pad 512K)
    const size_t OFF_QB   = OFF_WQKV + ((size_t)1 << 19);   // 1 MB
    const size_t OFF_KB   = OFF_QB   + ((size_t)1 << 20);   // 1 MB
    const size_t OFF_VT   = OFF_KB   + ((size_t)1 << 20);   // 1 MB
    const size_t OFF_SC   = OFF_VT   + ((size_t)1 << 20);   // 32 MB (scores/P)
    const size_t OFF_OUTB = OFF_SC   + ((size_t)32 << 20);  // 1 MB
    const size_t OFF_WLMT = OFF_OUTB + ((size_t)1 << 20);   // 4 MB

    unsigned short* xb   = (unsigned short*)(ws + OFF_XB);
    unsigned short* wt   = (unsigned short*)(ws + OFF_WQKV);
    unsigned short* qb   = (unsigned short*)(ws + OFF_QB);
    unsigned short* kbuf = (unsigned short*)(ws + OFF_KB);
    unsigned short* vt   = (unsigned short*)(ws + OFF_VT);
    unsigned short* scp  = (unsigned short*)(ws + OFF_SC);
    unsigned short* outb = (unsigned short*)(ws + OFF_OUTB);
    unsigned short* wlmt = (unsigned short*)(ws + OFF_WLMT);

    gather_cvt_kernel<<<ROWS, 256, 0, stream>>>(idx, emb, xb);
    cvt_wqkv_kernel<<<192, 256, 0, stream>>>(Wq, Wk, Wv, wt);
    qkv_wmma_kernel<<<dim3(12, 64), 256, 0, stream>>>(xb, wt, qb, kbuf, vt);
    scores_wmma_kernel<<<dim3(TLEN / 128, TLEN / 16, BATCH), 256, 0, stream>>>(qb, kbuf, scp);
    softmax_kernel<<<ROWS, 256, 0, stream>>>(scp);
    pv_wmma_kernel<<<64, 256, 0, stream>>>(scp, vt, outb);
    cvt_wlm_kernel<<<VOCAB / 256, 256, 0, stream>>>(Wlm, wlmt);
    logits_wmma_kernel<<<dim3(VOCAB / 256, ROWS / 16), 256, 0, stream>>>(outb, wlmt, b_lm, logits);
}
